// RelationAwareMultiHeadAttention_40149354283643
// MI455X (gfx1250) — matrix operand order repaired and checked by reading, not compiled, not run on hardware
//
#include <hip/hip_runtime.h>

typedef __attribute__((ext_vector_type(16))) _Float16 v16h;
typedef __attribute__((ext_vector_type(8)))  float    v8f;

#define SNB 136   // LDS row stride in halves (272B = 17*16B: aligned, conflict-free)

union V16HU { v16h v; uint4 q[2]; };

__global__ __launch_bounds__(256) void rmha_kernel(
    const float* __restrict__ drug,   // [B,128]
    const float* __restrict__ nbr,    // [B,128,128]
    const float* __restrict__ rel,    // [B,128,128]
    const float* __restrict__ Wq,     // [128,128]
    const float* __restrict__ Wk,     // [128,128]
    const float* __restrict__ Wv,     // [128,128]
    const float* __restrict__ Wr,     // [128,128]
    const float* __restrict__ Wo,     // [128,128]
    const float* __restrict__ bo,     // [128]
    const float* __restrict__ gamma,  // [128]
    const float* __restrict__ beta,   // [128]
    float* __restrict__ out)          // [B,128]
{
    __shared__ __align__(16) _Float16 s_mat[128 * SNB];  // nb[b] as f16
    __shared__ __align__(16) _Float16 s_u[2 * 4 * 128];  // u_k | u_r per head, f16
    __shared__ __align__(16) float s_drug[128];
    __shared__ __align__(16) float s_q[128];
    __shared__ __align__(16) float s_scores[4 * 128];
    __shared__ __align__(16) float s_c[4 * 128];         // context vectors
    __shared__ __align__(16) float s_agg[128];
    __shared__ __align__(16) float s_x[128];
    __shared__ float s_stat[2];

    const int t    = threadIdx.x;
    const int b    = blockIdx.x;
    const int wave = t >> 5;
    const int lane = t & 31;
    const int m    = lane & 15;   // A row / B column / C column within tile
    const int g    = lane >> 4;   // half-wave group

    // ---- stage neighbor_vecs[b] -> LDS f16; load drug; zero context ----
    {
        const float4* src = (const float4*)(nbr + (size_t)b * 16384);
        #pragma unroll
        for (int i = 0; i < 16; ++i) {
            int i4 = t + 256 * i;
            float4 f = src[i4];
            int e = i4 * 4, n = e >> 7, k = e & 127;
            _Float16* p = &s_mat[n * SNB + k];
            p[0] = (_Float16)f.x; p[1] = (_Float16)f.y;
            p[2] = (_Float16)f.z; p[3] = (_Float16)f.w;
        }
    }
    if (t < 128) s_drug[t] = drug[(size_t)b * 128 + t];
    for (int i = t; i < 512; i += 256) s_c[i] = 0.f;
    __syncthreads();

    // ---- q = Wq drug, pre-scaled by 1/sqrt(32) ----
    if (t < 128) {
        const float4* wr = (const float4*)(Wq + t * 128);
        const float4* dr = (const float4*)s_drug;
        float acc = 0.f;
        #pragma unroll 8
        for (int k = 0; k < 32; ++k) {
            float4 a = wr[k], d = dr[k];
            acc += a.x * d.x + a.y * d.y + a.z * d.z + a.w * d.w;
        }
        s_q[t] = acc * 0.17677669529663687f;
    }
    __syncthreads();

    // ---- u_k[h] = Wk^T q_h, u_r[h] = Wr^T q_h  (fp32, coalesced columns) ----
    {
        const int k = t & 127;
        const float* __restrict__ W = (t < 128) ? Wk : Wr;
        float u0 = 0.f, u1 = 0.f, u2 = 0.f, u3 = 0.f;
        for (int j = 0; j < 32; ++j) {           // coalesced across threads at each j
            u0 += W[(j      ) * 128 + k] * s_q[j      ];
            u1 += W[(j +  32) * 128 + k] * s_q[j +  32];
            u2 += W[(j +  64) * 128 + k] * s_q[j +  64];
            u3 += W[(j +  96) * 128 + k] * s_q[j +  96];
        }
        const int base = (t < 128) ? 0 : 512;
        s_u[base +   0 + k] = (_Float16)u0;
        s_u[base + 128 + k] = (_Float16)u1;
        s_u[base + 256 + k] = (_Float16)u2;
        s_u[base + 384 + k] = (_Float16)u3;
    }
    __syncthreads();

    // ---- scores via WMMA: [nb | rel](16x256 strip) @ [U_k ; U_r](256x16) ----
    // C columns 0..3 are heads; columns 4..15 are ignored junk (independent).
    {
        const int n0 = wave * 16;                // each wave owns a 16-row strip
        const float* relrow = rel + (size_t)b * 16384 + (size_t)(n0 + m) * 128;
        v8f acc = {};
        #pragma unroll
        for (int kk = 0; kk < 4; ++kk) {
            // A: nb strip from LDS (ISA 16x32 f16 layout)
            V16HU A;
            const _Float16* pa = &s_mat[(n0 + m) * SNB + kk * 32 + 8 * g];
            A.q[0] = *(const uint4*)pa;
            A.q[1] = *(const uint4*)(pa + 16);
            // B: U_k columns (lane m -> head m&3; junk lanes harmless)
            V16HU Bk;
            const _Float16* pb = &s_u[(m & 3) * 128 + kk * 32 + 8 * g];
            Bk.q[0] = *(const uint4*)pb;
            Bk.q[1] = *(const uint4*)(pb + 16);
            acc = __builtin_amdgcn_wmma_f32_16x16x32_f16(
                false, A.v, false, Bk.v, (short)0, acc, false, false);

            // A: rel strip streamed from global, f32 -> f16
            V16HU Ar;
            const float* pr0 = relrow + kk * 32 + 8 * g;
            const float* pr1 = relrow + kk * 32 + 16 + 8 * g;
            float4 f0 = *(const float4*)pr0, f1 = *(const float4*)(pr0 + 4);
            float4 f2 = *(const float4*)pr1, f3 = *(const float4*)(pr1 + 4);
            Ar.v[0]  = (_Float16)f0.x; Ar.v[1]  = (_Float16)f0.y;
            Ar.v[2]  = (_Float16)f0.z; Ar.v[3]  = (_Float16)f0.w;
            Ar.v[4]  = (_Float16)f1.x; Ar.v[5]  = (_Float16)f1.y;
            Ar.v[6]  = (_Float16)f1.z; Ar.v[7]  = (_Float16)f1.w;
            Ar.v[8]  = (_Float16)f2.x; Ar.v[9]  = (_Float16)f2.y;
            Ar.v[10] = (_Float16)f2.z; Ar.v[11] = (_Float16)f2.w;
            Ar.v[12] = (_Float16)f3.x; Ar.v[13] = (_Float16)f3.y;
            Ar.v[14] = (_Float16)f3.z; Ar.v[15] = (_Float16)f3.w;
            // B: U_r columns
            V16HU Br;
            const _Float16* pc = &s_u[512 + (m & 3) * 128 + kk * 32 + 8 * g];
            Br.q[0] = *(const uint4*)pc;
            Br.q[1] = *(const uint4*)(pc + 16);
            acc = __builtin_amdgcn_wmma_f32_16x16x32_f16(
                false, Ar.v, false, Br.v, (short)0, acc, false, false);
        }
        // C layout: lane holds column m (= head), rows n0 + 8g + r  -> direct store
        if (m < 4) {
            #pragma unroll
            for (int r = 0; r < 8; ++r)
                s_scores[m * 128 + n0 + 8 * g + r] = acc[r];
        }
    }
    __syncthreads();

    // ---- softmax over N per head (waves 0..3) ----
    if (wave < 4) {
        float v0 = s_scores[wave * 128 + lane];
        float v1 = s_scores[wave * 128 + lane + 32];
        float v2 = s_scores[wave * 128 + lane + 64];
        float v3 = s_scores[wave * 128 + lane + 96];
        float mx = fmaxf(fmaxf(v0, v1), fmaxf(v2, v3));
        for (int s = 1; s < 32; s <<= 1) mx = fmaxf(mx, __shfl_xor(mx, s, 32));
        float e0 = __expf(v0 - mx), e1 = __expf(v1 - mx);
        float e2 = __expf(v2 - mx), e3 = __expf(v3 - mx);
        float sm = e0 + e1 + e2 + e3;
        for (int s = 1; s < 32; s <<= 1) sm += __shfl_xor(sm, s, 32);
        float inv = 1.f / sm;
        s_scores[wave * 128 + lane]      = e0 * inv;
        s_scores[wave * 128 + lane + 32] = e1 * inv;
        s_scores[wave * 128 + lane + 64] = e2 * inv;
        s_scores[wave * 128 + lane + 96] = e3 * inv;
    }
    __syncthreads();

    // ---- context vectors: c[h] = sum_n alpha[h][n] * nb[n]  (from LDS) ----
    {
        const int k  = t & 127;
        const int nb0 = (t >> 7) * 64;
        float c0 = 0.f, c1 = 0.f, c2 = 0.f, c3 = 0.f;
        for (int n = nb0; n < nb0 + 64; ++n) {
            float x = (float)s_mat[n * SNB + k];
            c0 += s_scores[n      ] * x;
            c1 += s_scores[n + 128] * x;
            c2 += s_scores[n + 256] * x;
            c3 += s_scores[n + 384] * x;
        }
        atomicAdd(&s_c[k      ], c0);
        atomicAdd(&s_c[k + 128], c1);
        atomicAdd(&s_c[k + 256], c2);
        atomicAdd(&s_c[k + 384], c3);
    }
    __syncthreads();

    // ---- agg[j] = <c[j/32], Wv row j> ----
    if (t < 128) {
        const float4* wv = (const float4*)(Wv + t * 128);
        const float4* cc = (const float4*)(&s_c[(t >> 5) * 128]);
        float acc = 0.f;
        #pragma unroll 8
        for (int k = 0; k < 32; ++k) {
            float4 a = wv[k], d = cc[k];
            acc += a.x * d.x + a.y * d.y + a.z * d.z + a.w * d.w;
        }
        s_agg[t] = acc;
    }
    __syncthreads();

    // ---- output projection + residual ----
    if (t < 128) {
        const float4* wo = (const float4*)(Wo + t * 128);
        const float4* ag = (const float4*)s_agg;
        float acc = bo[t] + s_drug[t];
        #pragma unroll 8
        for (int k = 0; k < 32; ++k) {
            float4 a = wo[k], d = ag[k];
            acc += a.x * d.x + a.y * d.y + a.z * d.z + a.w * d.w;
        }
        s_x[t] = acc;
    }
    __syncthreads();

    // ---- LayerNorm stats (wave 0) ----
    if (t < 32) {
        float s = 0.f, s2 = 0.f;
        for (int k = lane; k < 128; k += 32) { float v = s_x[k]; s += v; s2 += v * v; }
        for (int sh = 1; sh < 32; sh <<= 1) {
            s  += __shfl_xor(s,  sh, 32);
            s2 += __shfl_xor(s2, sh, 32);
        }
        if (lane == 0) {
            float mu  = s * (1.f / 128.f);
            float var = s2 * (1.f / 128.f) - mu * mu;
            s_stat[0] = mu;
            s_stat[1] = rsqrtf(var + 1e-5f);
        }
    }
    __syncthreads();

    if (t < 128)
        out[(size_t)b * 128 + t] =
            (s_x[t] - s_stat[0]) * s_stat[1] * gamma[t] + beta[t];
}

extern "C" void kernel_launch(void* const* d_in, const int* in_sizes, int n_in,
                              void* d_out, int out_size, void* d_ws, size_t ws_size,
                              hipStream_t stream) {
    const float* drug  = (const float*)d_in[0];
    const float* nbr   = (const float*)d_in[1];
    const float* rel   = (const float*)d_in[2];
    const float* Wq    = (const float*)d_in[3];
    const float* Wk    = (const float*)d_in[4];
    const float* Wv    = (const float*)d_in[5];
    const float* Wr    = (const float*)d_in[6];
    const float* Wo    = (const float*)d_in[7];
    const float* bo    = (const float*)d_in[8];
    const float* gamma = (const float*)d_in[9];
    const float* beta  = (const float*)d_in[10];
    float* out = (float*)d_out;

    const int B = in_sizes[0] / 128;   // 4096

    rmha_kernel<<<B, 256, 0, stream>>>(drug, nbr, rel, Wq, Wk, Wv, Wr, Wo,
                                       bo, gamma, beta, out);
}
